// SupernovaAttention_63393717289471
// MI455X (gfx1250) — matrix operand order repaired and checked by reading, not compiled, not run on hardware
//
#include <hip/hip_runtime.h>
#include <hip/hip_bf16.h>
#include <stdint.h>

typedef __attribute__((ext_vector_type(16))) _Float16 v16h;
typedef __attribute__((ext_vector_type(8)))  _Float16 v8h;
typedef __attribute__((ext_vector_type(4)))  _Float16 v4h;
typedef __attribute__((ext_vector_type(2)))  _Float16 v2h;
typedef __attribute__((ext_vector_type(8)))  float    v8f;

namespace sup {

constexpr int B   = 2;
constexpr int S   = 2048;
constexpr int HID = 2048;
constexpr int NH  = 16;
constexpr int NKV = 4;
constexpr int HD  = 128;
constexpr int WIN = 512;
constexpr float QSCALE = 0.08838834764831845f;   // 1/sqrt(128)
constexpr float NEGBIG = -1e30f;

// ---------------- CDNA5 async global->LDS copy (ASYNCcnt-tracked) ----------------
__device__ __forceinline__ void g2l_b128(void* lds, const void* gsrc) {
  asm volatile("global_load_async_to_lds_b128 %0, %1, off"
               :: "v"((uint32_t)(uintptr_t)lds), "v"(gsrc)
               : "memory");
}
__device__ __forceinline__ void wait_async() {
  asm volatile("s_wait_asynccnt 0x0" ::: "memory");
}

// ---------------- DPP ROW_XMASK 16-lane reductions (no LDS traffic) ----------------
template <int CTRL>
__device__ __forceinline__ float dpp_f(float v) {
  return __int_as_float(
      __builtin_amdgcn_update_dpp(0, __float_as_int(v), CTRL, 0xf, 0xf, true));
}
__device__ __forceinline__ float red_max16(float v) {
  v = fmaxf(v, dpp_f<0x161>(v));   // ROW_XMASK:1
  v = fmaxf(v, dpp_f<0x162>(v));   // ROW_XMASK:2
  v = fmaxf(v, dpp_f<0x164>(v));   // ROW_XMASK:4
  v = fmaxf(v, dpp_f<0x168>(v));   // ROW_XMASK:8
  return v;
}
__device__ __forceinline__ float red_add16(float v) {
  v += dpp_f<0x161>(v);
  v += dpp_f<0x162>(v);
  v += dpp_f<0x164>(v);
  v += dpp_f<0x168>(v);
  return v;
}

// Build a v16h fragment from two aligned 16B LDS/global chunks.
union frag16 { v16h v; v8h h[2]; };
__device__ __forceinline__ v16h ldfrag(const _Float16* p0, const _Float16* p1) {
  frag16 f;
  f.h[0] = *(const v8h*)p0;
  f.h[1] = *(const v8h*)p1;
  return f.v;
}

__device__ __forceinline__ v8f wmma_f16(v16h a, v16h b, v8f c) {
  return __builtin_amdgcn_wmma_f32_16x16x32_f16(false, a, false, b, (short)0, c,
                                                false, false);
}

// ---------------- fp32 -> fp16 convert, 4 elements / thread ----------------
__global__ __launch_bounds__(256) void cvt_f32_to_f16(const float* __restrict__ src,
                                                      _Float16* __restrict__ dst) {
  size_t i = ((size_t)blockIdx.x * 256 + threadIdx.x) * 4;
  const float4 v = *(const float4*)(src + i);
  v4h o = { (_Float16)v.x, (_Float16)v.y, (_Float16)v.z, (_Float16)v.w };
  *(v4h*)(dst + i) = o;
}

// ---------------- WMMA GEMM: C[M,N] = A[M,K] * W[N,K]^T ----------------
// 128x128 block, BK=32, 256 threads = 8 waves, wave -> 64Mx32N (4x2 WMMA tiles).
// Double-buffered LDS: async copy of tile k+1 overlapped with WMMA on tile k.
template <bool F16OUT>
__global__ __launch_bounds__(256) void gemm_wmma_kernel(const _Float16* __restrict__ A,
                                                        const _Float16* __restrict__ W,
                                                        void* __restrict__ Cout,
                                                        int M, int N, int Kd) {
  __shared__ _Float16 As[2][128 * 32];
  __shared__ _Float16 Ws[2][128 * 32];

  const int tid  = threadIdx.x;
  const int lane = tid & 31;
  const int wv   = tid >> 5;
  const int half = lane >> 4;
  const int lm   = lane & 15;
  const int m0 = blockIdx.y * 128;
  const int n0 = blockIdx.x * 128;
  const int wm = (wv & 1) * 64;
  const int wn = (wv >> 1) * 32;

  auto stage = [&](int buf, int k0) {
#pragma unroll
    for (int i = 0; i < 2; i++) {
      int c = tid + i * 256;          // 512 16B chunks per tile
      int rr = c >> 2;                // row 0..127
      int ch = (c & 3) * 8;
      g2l_b128(&As[buf][rr * 32 + ch], A + (size_t)(m0 + rr) * Kd + k0 + ch);
      g2l_b128(&Ws[buf][rr * 32 + ch], W + (size_t)(n0 + rr) * Kd + k0 + ch);
    }
  };

  v8f acc[4][2];
#pragma unroll
  for (int mi = 0; mi < 4; mi++)
#pragma unroll
    for (int ni = 0; ni < 2; ni++) acc[mi][ni] = {};

  stage(0, 0);
  int cur = 0;
  for (int k0 = 0; k0 < Kd; k0 += 32) {
    wait_async();        // tile `cur` resident
    __syncthreads();     // all waves done writing cur / done reading cur^1
    if (k0 + 32 < Kd) stage(cur ^ 1, k0 + 32);  // overlap next copy with compute

    v16h af[4], bf[2];
#pragma unroll
    for (int mi = 0; mi < 4; mi++) {  // A frag: row per lane, K-halves by lane group
      const _Float16* p = &As[cur][(wm + mi * 16 + lm) * 32 + half * 8];
      af[mi] = ldfrag(p, p + 16);
    }
#pragma unroll
    for (int ni = 0; ni < 2; ni++) {  // B frag: column per lane, K-halves by lane group
      const _Float16* p = &Ws[cur][(wn + ni * 16 + lm) * 32 + half * 8];
      bf[ni] = ldfrag(p, p + 16);
    }
#pragma unroll
    for (int mi = 0; mi < 4; mi++)
#pragma unroll
      for (int ni = 0; ni < 2; ni++)
        acc[mi][ni] = wmma_f16(af[mi], bf[ni], acc[mi][ni]);
    cur ^= 1;
  }

#pragma unroll
  for (int mi = 0; mi < 4; mi++)
#pragma unroll
    for (int ni = 0; ni < 2; ni++)
#pragma unroll
      for (int e = 0; e < 8; e++) {
        int row = m0 + wm + mi * 16 + half * 8 + e;
        int col = n0 + wn + ni * 16 + lm;
        if (F16OUT)
          ((_Float16*)Cout)[(size_t)row * N + col] = (_Float16)acc[mi][ni][e];
        else
          ((float*)Cout)[(size_t)row * N + col] = acc[mi][ni][e];
      }
}

// ---------------- RoPE (+ optional scale folded into Q) ----------------
__global__ __launch_bounds__(256) void rope_scale_kernel(_Float16* __restrict__ X,
                                                         int nheads, float scale) {
  int i   = blockIdx.x * 256 + threadIdx.x;
  int per = nheads * 32;
  int bs  = i / per;
  int r   = i % per;
  int h   = r >> 5;
  int f   = r & 31;
  int s   = bs & (S - 1);
  float invf = __expf(-(float)f * (9.210340371976184f / 32.0f));  // 10000^(-f/32)
  float ang  = (float)s * invf;
  float sn, cs;
  __sincosf(ang, &sn, &cs);
  _Float16* p = X + (size_t)bs * (nheads * HD) + h * HD;
  float x1 = (float)p[f];
  float x2 = (float)p[f + 32];
  p[f]      = (_Float16)((x1 * cs - x2 * sn) * scale);
  p[f + 32] = (_Float16)((x2 * cs + x1 * sn) * scale);
  p[f + 64] = (_Float16)((float)p[f + 64] * scale);
  p[f + 96] = (_Float16)((float)p[f + 96] * scale);
}

// ---------------- Sliding-window flash attention ----------------
// 128 threads = 4 waves; block = 64 q rows of one (b,h); wave -> 16 rows.
// Double-buffered K (async copy) and V^T (load-early/store-late) tiles.
__global__ __launch_bounds__(128) void attn_kernel(const _Float16* __restrict__ Qg,
                                                   const _Float16* __restrict__ Kg,
                                                   const _Float16* __restrict__ Vg,
                                                   _Float16* __restrict__ Oa) {
  __shared__ _Float16 Kt[2][32 * 128];   // [key][d]
  __shared__ _Float16 Vt[2][128 * 32];   // [d][key] (transposed for B-frag)
  __shared__ _Float16 Pt[4][16 * 32];

  const int bh   = blockIdx.y;
  const int b    = bh >> 4;          // NH = 16
  const int h    = bh & 15;
  const int kvh  = h >> 2;           // n_rep = 4
  const int q0   = blockIdx.x * 64;
  const int tid  = threadIdx.x;
  const int wave = tid >> 5;
  const int lane = tid & 31;
  const int half = lane >> 4;
  const int lm   = lane & 15;
  const int qw   = q0 + wave * 16;

  auto stageK = [&](int buf, int js) {
#pragma unroll
    for (int i = 0; i < 4; i++) {
      int c = tid + i * 128;          // 512 16B chunks
      int rr = c >> 4;
      int ch = (c & 15) * 8;
      g2l_b128(&Kt[buf][rr * 128 + ch],
               Kg + (size_t)(b * S + js + rr) * (NKV * HD) + kvh * HD + ch);
    }
  };
  const int vj = (tid & 15) * 2;      // key pair owned by this thread
  const int vd = (tid >> 4) * 16;     // d block owned by this thread
  auto loadV = [&](int js, v8h* t) {
    const _Float16* vs = Vg + (size_t)(b * S + js + vj) * (NKV * HD) + kvh * HD + vd;
    t[0] = *(const v8h*)(vs);
    t[1] = *(const v8h*)(vs + 8);
    t[2] = *(const v8h*)(vs + NKV * HD);
    t[3] = *(const v8h*)(vs + NKV * HD + 8);
  };
  auto storeVt = [&](int buf, const v8h* t) {
#pragma unroll
    for (int i = 0; i < 16; i++) {    // paired-key b32 stores
      v2h p2 = { t[i >> 3][i & 7], t[2 + (i >> 3)][i & 7] };
      *(v2h*)&Vt[buf][(vd + i) * 32 + vj] = p2;
    }
  };

  // Q fragments (A-layout), HD=128 -> 4 x K32 frags, held in registers
  v16h qf[4];
  {
    const _Float16* qb = Qg + (size_t)(b * S + qw + lm) * (NH * HD) + h * HD;
#pragma unroll
    for (int c = 0; c < 4; c++) {
      const _Float16* p = qb + c * 32 + half * 8;
      qf[c] = ldfrag(p, p + 16);
    }
  }

  v8f o[8];
#pragma unroll
  for (int n = 0; n < 8; n++) o[n] = {};
  float ms[8], ls[8];
#pragma unroll
  for (int e = 0; e < 8; e++) { ms[e] = NEGBIG; ls[e] = 0.f; }

  int kstart = q0 - WIN + 1;
  if (kstart < 0) kstart = 0;
  kstart &= ~31;

  _Float16* Pw = Pt[wave];
  v8h vr[4];

  stageK(0, kstart);
  loadV(kstart, vr);
  storeVt(0, vr);
  int cur = 0;

  for (int js = kstart; js <= q0 + 63; js += 32) {
    wait_async();
    __syncthreads();
    const bool more = (js + 32 <= q0 + 63);
    if (more) {
      stageK(cur ^ 1, js + 32);   // async copy overlapped with compute below
      loadV(js + 32, vr);         // global loads issued now, consumed after compute
    }

    // Two 16-key subtiles: scores + online softmax (DPP reductions)
#pragma unroll
    for (int nt = 0; nt < 2; nt++) {
      v8f sc8 = {};
#pragma unroll
      for (int c = 0; c < 4; c++) {  // contraction over HD
        const _Float16* kp = &Kt[cur][(nt * 16 + lm) * 128 + c * 32 + half * 8];
        sc8 = wmma_f16(qf[c], ldfrag(kp, kp + 16), sc8);
      }
      const int j = js + nt * 16 + lm;
#pragma unroll
      for (int e = 0; e < 8; e++) {
        const int qr = qw + half * 8 + e;
        float val  = ((j <= qr) && (j > qr - WIN)) ? sc8[e] : NEGBIG;
        float rmax = red_max16(val);
        float mn   = fmaxf(ms[e], rmax);
        float rsc  = __expf(ms[e] - mn);
        float p    = __expf(val - mn);
        float rs   = red_add16(p);
        ls[e] = ls[e] * rsc + rs;
        ms[e] = mn;
#pragma unroll
        for (int n = 0; n < 8; n++) o[n][e] *= rsc;
        Pw[(half * 8 + e) * 32 + nt * 16 + lm] = (_Float16)p;
      }
    }

    // P(16x32, A-layout via per-wave LDS bounce) x V^T -> O accumulate
    const _Float16* pp = Pw + lm * 32 + half * 8;
    v16h pf = ldfrag(pp, pp + 16);
#pragma unroll
    for (int n = 0; n < 8; n++) {
      const _Float16* vp = &Vt[cur][(n * 16 + lm) * 32 + half * 8];
      o[n] = wmma_f16(pf, ldfrag(vp, vp + 16), o[n]);
    }

    if (more) storeVt(cur ^ 1, vr);  // LDS stores after compute; barrier next iter
    cur ^= 1;
  }

  // normalize + store
  float inv[8];
#pragma unroll
  for (int e = 0; e < 8; e++) inv[e] = 1.0f / ls[e];
  _Float16* ob = Oa + (size_t)(b * S + qw + half * 8) * (NH * HD) + h * HD + lm;
#pragma unroll
  for (int n = 0; n < 8; n++)
#pragma unroll
    for (int e = 0; e < 8; e++)
      ob[(size_t)e * (NH * HD) + n * 16] = (_Float16)(o[n][e] * inv[e]);
}

}  // namespace sup

extern "C" void kernel_launch(void* const* d_in, const int* in_sizes, int n_in,
                              void* d_out, int out_size, void* d_ws, size_t ws_size,
                              hipStream_t stream) {
  using namespace sup;
  const float* hs = (const float*)d_in[0];
  const float* Wq = (const float*)d_in[1];
  const float* Wk = (const float*)d_in[2];
  const float* Wv = (const float*)d_in[3];
  const float* Wo = (const float*)d_in[4];
  float* out = (float*)d_out;

  constexpr size_t MEG = 1024 * 1024;
  _Float16* h16  = (_Float16*)d_ws;        // 8M elems
  _Float16* q16  = h16  + 8 * MEG;         // 8M
  _Float16* a16  = q16  + 8 * MEG;         // 8M (attention output)
  _Float16* wq16 = a16  + 8 * MEG;         // 4M
  _Float16* wo16 = wq16 + 4 * MEG;         // 4M
  _Float16* wk16 = wo16 + 4 * MEG;         // 1M
  _Float16* wv16 = wk16 + 1 * MEG;         // 1M
  _Float16* k16  = wv16 + 1 * MEG;         // 2M
  _Float16* v16  = k16  + 2 * MEG;         // 2M

  // fp32 -> fp16 staging (sizes all divisible by 1024)
  cvt_f32_to_f16<<<(B * S * HID) / 1024, 256, 0, stream>>>(hs, h16);
  cvt_f32_to_f16<<<(NH * HD * HID) / 1024, 256, 0, stream>>>(Wq, wq16);
  cvt_f32_to_f16<<<(NKV * HD * HID) / 1024, 256, 0, stream>>>(Wk, wk16);
  cvt_f32_to_f16<<<(NKV * HD * HID) / 1024, 256, 0, stream>>>(Wv, wv16);
  cvt_f32_to_f16<<<(HID * NH * HD) / 1024, 256, 0, stream>>>(Wo, wo16);

  const int M = B * S;  // 4096
  // QKV projections (WMMA, f16 out)
  gemm_wmma_kernel<true><<<dim3((NH * HD) / 128, M / 128), 256, 0, stream>>>(
      h16, wq16, q16, M, NH * HD, HID);
  gemm_wmma_kernel<true><<<dim3((NKV * HD) / 128, M / 128), 256, 0, stream>>>(
      h16, wk16, k16, M, NKV * HD, HID);
  gemm_wmma_kernel<true><<<dim3((NKV * HD) / 128, M / 128), 256, 0, stream>>>(
      h16, wv16, v16, M, NKV * HD, HID);

  // RoPE: Q gets 1/sqrt(HD) folded in; K unscaled
  rope_scale_kernel<<<(M * NH * 32) / 256, 256, 0, stream>>>(q16, NH, QSCALE);
  rope_scale_kernel<<<(M * NKV * 32) / 256, 256, 0, stream>>>(k16, NKV, 1.0f);

  // Sliding-window attention
  attn_kernel<<<dim3(S / 64, B * NH), 128, 0, stream>>>(q16, k16, v16, a16);

  // Output projection (f32 out)
  gemm_wmma_kernel<false><<<dim3(HID / 128, M / 128), 256, 0, stream>>>(
      a16, wo16, out, M, HID, NH * HD);
}
